// Encode_61469571940433
// MI455X (gfx1250) — matrix operand order repaired and checked by reading, not compiled, not run on hardware
//
#include <hip/hip_runtime.h>
#include <hip/hip_bf16.h>

// MI455X / gfx1250: bf16 WMMA for all four GEMMs + flash attention core.
// V-block staging in attention goes through the Tensor Data Mover
// (tensor_load_to_lds + s_wait_tensorcnt) when the builtin is available.
// Workspace layout (bf16): q[2*1024*1024] | k[2*4096*1024] | v[2*4096*1024] | attn[2*1024*1024]
// = 40 MiB total.

#define DIMW 1024
#define NHEADS 16
#define HD 64
#define NLAT 1024
#define MTOK 4096

typedef __bf16 bf16_t;
typedef __attribute__((ext_vector_type(16))) __bf16 v16bf;
typedef __attribute__((ext_vector_type(8)))  float  v8f;

typedef unsigned int u32x4 __attribute__((ext_vector_type(4)));
typedef int          i32x4 __attribute__((ext_vector_type(4)));
typedef int          i32x8 __attribute__((ext_vector_type(8)));

#if __has_builtin(__builtin_amdgcn_tensor_load_to_lds) && \
    __has_builtin(__builtin_amdgcn_s_wait_tensorcnt)
#define USE_TDM 1
#else
#define USE_TDM 0
#endif

union V16U { v16bf v; uint4 u[2]; bf16_t b[16]; };
union BF8U { uint4 u; bf16_t b[8]; };

__device__ __forceinline__ v16bf ld_frag_2x16B(const bf16_t* p0, const bf16_t* p1) {
  V16U r;
  r.u[0] = *(const uint4*)p0;
  r.u[1] = *(const uint4*)p1;
  return r.v;
}
__device__ __forceinline__ v16bf ld_frag_32B(const bf16_t* p) {
  V16U r;
  r.u[0] = *(const uint4*)(p + 0);
  r.u[1] = *(const uint4*)(p + 8);
  return r.v;
}
__device__ __forceinline__ uint4 cvt8_bf16(float4 f0, float4 f1) {
  BF8U p;
  p.b[0] = (bf16_t)f0.x; p.b[1] = (bf16_t)f0.y; p.b[2] = (bf16_t)f0.z; p.b[3] = (bf16_t)f0.w;
  p.b[4] = (bf16_t)f1.x; p.b[5] = (bf16_t)f1.y; p.b[6] = (bf16_t)f1.z; p.b[7] = (bf16_t)f1.w;
  return p.u;
}

// ---------------------------------------------------------------------------
// GEMM: Y[r][c] = sum_k A[r][k] * W[c][k] + bias[c]   (A @ W^T + b)
// Block: 256 rows x 64 cols, 256 threads = 8 waves; each wave: 32 rows x 64 cols
// (two A-fragments share each B-fragment -> 8 WMMAs per barrier pair).
// ---------------------------------------------------------------------------
template<bool A_F32, bool OUT_F32>
__global__ __launch_bounds__(256) void gemm_bias_wmma(
    const float* __restrict__ A32, const bf16_t* __restrict__ A16,
    const float* __restrict__ W, const float* __restrict__ bias,
    float* __restrict__ O32, bf16_t* __restrict__ O16)
{
  __shared__ __align__(16) bf16_t As[256][32];
  __shared__ __align__(16) bf16_t Ws[64][32];

  const int tid  = threadIdx.x;
  const int wave = tid >> 5;
  const int lane = tid & 31;
  const int hi   = lane >> 4;
  const int ln   = lane & 15;
  const int r0 = blockIdx.x * 256;
  const int c0 = blockIdx.y * 64;

  const v8f vzero = {0.f,0.f,0.f,0.f,0.f,0.f,0.f,0.f};
  v8f acc[2][4];
#pragma unroll
  for (int rt = 0; rt < 2; ++rt)
#pragma unroll
    for (int nt = 0; nt < 4; ++nt) acc[rt][nt] = vzero;

  const int wr = tid >> 2;            // 0..63
  const int wk = (tid & 3) * 8;       // 0,8,16,24

  for (int kk = 0; kk < DIMW; kk += 32) {
    // --- stage A tile (256 rows x 32 k) as bf16; one full row strip per thread ---
    if (A_F32) {
      const float* src = A32 + (size_t)(r0 + tid) * DIMW + kk;
      __builtin_prefetch(src + 32, 0, 1);   // next K-tile -> global_prefetch_b8
      float4 f0 = *(const float4*)(src + 0);
      float4 f1 = *(const float4*)(src + 4);
      float4 f2 = *(const float4*)(src + 8);
      float4 f3 = *(const float4*)(src + 12);
      float4 f4 = *(const float4*)(src + 16);
      float4 f5 = *(const float4*)(src + 20);
      float4 f6 = *(const float4*)(src + 24);
      float4 f7 = *(const float4*)(src + 28);
      *(uint4*)&As[tid][0]  = cvt8_bf16(f0, f1);
      *(uint4*)&As[tid][8]  = cvt8_bf16(f2, f3);
      *(uint4*)&As[tid][16] = cvt8_bf16(f4, f5);
      *(uint4*)&As[tid][24] = cvt8_bf16(f6, f7);
    } else {
      const bf16_t* src = A16 + (size_t)(r0 + tid) * DIMW + kk;
      __builtin_prefetch(src + 32, 0, 1);
      *(uint4*)&As[tid][0]  = *(const uint4*)(src + 0);
      *(uint4*)&As[tid][8]  = *(const uint4*)(src + 8);
      *(uint4*)&As[tid][16] = *(const uint4*)(src + 16);
      *(uint4*)&As[tid][24] = *(const uint4*)(src + 24);
    }
    // --- stage W tile (64 output cols x 32 k) as bf16 ---
    {
      const float* src = W + (size_t)(c0 + wr) * DIMW + kk + wk;
      __builtin_prefetch(src + 32, 0, 1);
      float4 f0 = *(const float4*)(src + 0);
      float4 f1 = *(const float4*)(src + 4);
      *(uint4*)&Ws[wr][wk] = cvt8_bf16(f0, f1);
    }
    __syncthreads();

    const int kb8  = hi * 8;
    const int kb16 = hi * 16;
    const int arow = wave * 32 + ln;
    v16bf a0 = ld_frag_2x16B(&As[arow][kb8],      &As[arow][kb8 + 16]);
    v16bf a1 = ld_frag_2x16B(&As[arow + 16][kb8], &As[arow + 16][kb8 + 16]);
#pragma unroll
    for (int nt = 0; nt < 4; ++nt) {
      v16bf bfr = ld_frag_2x16B(&Ws[ln + 16*nt][kb8], &Ws[ln + 16*nt][kb8 + 16]);   // B[k][n] = W[c0+n][k]
      acc[0][nt] = __builtin_amdgcn_wmma_f32_16x16x32_bf16(
          false, a0, false, bfr, (short)0, acc[0][nt], false, false);
      acc[1][nt] = __builtin_amdgcn_wmma_f32_16x16x32_bf16(
          false, a1, false, bfr, (short)0, acc[1][nt], false, false);
    }
    __syncthreads();
  }

  // --- epilogue: bias + store. C layout: elem vr -> row vr + 8*hi, col ln+16*nt ---
#pragma unroll
  for (int rt = 0; rt < 2; ++rt) {
#pragma unroll
    for (int nt = 0; nt < 4; ++nt) {
      const int col = c0 + ln + 16*nt;
      const float bv = bias[col];
#pragma unroll
      for (int vr = 0; vr < 8; ++vr) {
        const int row = r0 + wave*32 + rt*16 + vr + 8*hi;
        const float val = acc[rt][nt][vr] + bv;
        if (OUT_F32) O32[(size_t)row * DIMW + col] = val;
        else         O16[(size_t)row * DIMW + col] = (bf16_t)val;
      }
    }
  }
}

// ---------------------------------------------------------------------------
// Flash attention with block-causal mask (j//4 <= i).
// One wave32 per (b, h, 16-row Q tile). KV processed in 64-wide blocks
// (16 WMMAs per softmax round). KV extent = 4*i0 + 64 (multiple of 64).
// V block staged via Tensor Data Mover when available.
// ---------------------------------------------------------------------------
__global__ __launch_bounds__(32) void attn_flash_wmma(
    const bf16_t* __restrict__ Q, const bf16_t* __restrict__ K,
    const bf16_t* __restrict__ V, bf16_t* __restrict__ O)
{
  __shared__ __align__(16) bf16_t Vs[64][64];   // KV block (64 tokens x 64 d)
  __shared__ __align__(16) bf16_t Ps[16][64];   // P tile for C->A layout swap

  const int lane = threadIdx.x;
  const int hi = lane >> 4;
  const int ln = lane & 15;
  const int i0 = blockIdx.x * 16;
  const int h  = blockIdx.y;
  const int b  = blockIdx.z;

  const size_t qbase  = (size_t)b * NLAT * DIMW + (size_t)h * HD;
  const size_t kvbase = (size_t)b * MTOK * DIMW + (size_t)h * HD;
  const int kb8  = hi * 8;
  const int kb16 = hi * 16;

  // Q A-fragments for d = 0..31 and d = 32..63 (loaded once)
  v16bf qf0, qf1;
  {
    const bf16_t* qp = Q + qbase + (size_t)(i0 + ln) * DIMW;
    qf0 = ld_frag_2x16B(qp + kb8,      qp + kb8 + 16);
    qf1 = ld_frag_2x16B(qp + 32 + kb8, qp + 32 + kb8 + 16);
  }

  const v8f vzero = {0.f,0.f,0.f,0.f,0.f,0.f,0.f,0.f};
  v8f o[4];
  o[0] = vzero; o[1] = vzero; o[2] = vzero; o[3] = vzero;
  float runmax[8], runsum[8];
#pragma unroll
  for (int vr = 0; vr < 8; ++vr) { runmax[vr] = -3.0e38f; runsum[vr] = 0.f; }

  const float scale = 0.125f;           // 1/sqrt(64)
  const int jend = 4 * i0 + 64;         // block-causal KV extent, multiple of 64

  for (int j0 = 0; j0 < jend; j0 += 64) {
    // --- prefetch next K block (global_prefetch_b8; speculative, OOB-safe) ---
    __builtin_prefetch(K + kvbase + (size_t)(j0 + 64 + lane) * DIMW, 0, 1);

#if USE_TDM
    // --- stage V block (64 x 64 bf16) via Tensor Data Mover ---
    // D# group0: count=1 | lds_addr | 57-bit global addr | type=2 ("image")
    // D# group1: data_size=2B, tensor_dim0=1024, tensor_dim1=8192 rows,
    //            tile 64x64, tensor_dim0_stride=1024 elements.
    {
      const uint64_t ga = (uint64_t)(uintptr_t)(V + kvbase + (size_t)j0 * DIMW);
      const unsigned  la = (unsigned)(uintptr_t)&Vs[0][0];   // LDS aperture: low 32 bits
      u32x4 g0;
      g0[0] = 1u;                                            // count=1 valid descriptor
      g0[1] = la;                                            // lds_addr
      g0[2] = (unsigned)ga;                                  // global_addr[31:0]
      g0[3] = ((unsigned)(ga >> 32) & 0x01FFFFFFu) | (2u << 30);  // addr[56:32] | type=2
      i32x8 g1;
      g1[0] = 1 << 16;                 // workgroup_mask=0, data_size=1 (2 bytes)
      g1[1] = (int)(1024u << 16);      // tensor_dim0[15:0] = 1024  (bits 63:48)
      g1[2] = (int)(8192u << 16);      // tensor_dim0 hi = 0 ; tensor_dim1[15:0] = 8192
      g1[3] = (int)(64u << 16);        // tensor_dim1 hi = 0 ; tile_dim0 = 64
      g1[4] = 64;                      // tile_dim1 = 64 ; tile_dim2 = 0
      g1[5] = 1024;                    // tensor_dim0_stride[31:0] = 1024
      g1[6] = 0;                       // stride hi = 0 ; tensor_dim1_stride lo = 0
      g1[7] = 0;
      const i32x4 gz = {0, 0, 0, 0};   // groups 2/3 unused (2D tensor)
#if defined(__clang_major__) && __clang_major__ >= 23
      const i32x8 gz8 = {0, 0, 0, 0, 0, 0, 0, 0};
      __builtin_amdgcn_tensor_load_to_lds(g0, g1, gz, gz, gz8, 0);
#else
      __builtin_amdgcn_tensor_load_to_lds(g0, g1, gz, gz, 0);
#endif
    }
#else
    // --- fallback: stage V block manually, two token rows per lane ---
#pragma unroll
    for (int rr = 0; rr < 2; ++rr) {
      const int r = lane + 32*rr;
      const bf16_t* vp = V + kvbase + (size_t)(j0 + r) * DIMW;
#pragma unroll
      for (int c = 0; c < 4; ++c) {
        *(uint4*)&Vs[r][c*16 + 0] = *(const uint4*)(vp + c*16 + 0);
        *(uint4*)&Vs[r][c*16 + 8] = *(const uint4*)(vp + c*16 + 8);
      }
    }
#endif

    // --- S = Q @ K^T : four 16x16 N-tiles, K(d)=64 split into 2 WMMAs each ---
    v8f s[4];
#pragma unroll
    for (int nt = 0; nt < 4; ++nt) s[nt] = vzero;
#pragma unroll
    for (int nt = 0; nt < 4; ++nt) {
      const bf16_t* kp = K + kvbase + (size_t)(j0 + 16*nt + ln) * DIMW;
      v16bf bk0 = ld_frag_2x16B(kp + kb8, kp + kb8 + 16);        // B[k][n] = K[j][d]
      v16bf bk1 = ld_frag_2x16B(kp + 32 + kb8, kp + 32 + kb8 + 16);
      s[nt] = __builtin_amdgcn_wmma_f32_16x16x32_bf16(
          false, qf0, false, bk0, (short)0, s[nt], false, false);
      s[nt] = __builtin_amdgcn_wmma_f32_16x16x32_bf16(
          false, qf1, false, bk1, (short)0, s[nt], false, false);
    }

    // --- scale + block-causal mask + block row-max ---
    float bm[8];
#pragma unroll
    for (int vr = 0; vr < 8; ++vr) {
      const int irow = i0 + vr + 8*hi;
      const int lim = 4 * (irow + 1);       // allowed: j < lim
      float m = -3.0e38f;
#pragma unroll
      for (int nt = 0; nt < 4; ++nt) {
        const int jcol = j0 + 16*nt + ln;
        const float val = (jcol < lim) ? s[nt][vr] * scale : -3.0e38f;
        s[nt][vr] = val;
        m = fmaxf(m, val);
      }
      bm[vr] = m;
    }
#pragma unroll
    for (int off = 1; off < 16; off <<= 1) {
#pragma unroll
      for (int vr = 0; vr < 8; ++vr)
        bm[vr] = fmaxf(bm[vr], __shfl_xor(bm[vr], off, 32));
    }

    // --- online softmax update ---
    float alpha[8], bs[8];
#pragma unroll
    for (int vr = 0; vr < 8; ++vr) {
      const float nm = fmaxf(runmax[vr], bm[vr]);
      alpha[vr] = __expf(runmax[vr] - nm);
      runmax[vr] = nm;
      float acc_s = 0.f;
#pragma unroll
      for (int nt = 0; nt < 4; ++nt) {
        const float p = __expf(s[nt][vr] - nm);
        s[nt][vr] = p;
        acc_s += p;
      }
      bs[vr] = acc_s;
    }
#pragma unroll
    for (int off = 1; off < 16; off <<= 1) {
#pragma unroll
      for (int vr = 0; vr < 8; ++vr)
        bs[vr] += __shfl_xor(bs[vr], off, 32);
    }
#pragma unroll
    for (int vr = 0; vr < 8; ++vr)
      runsum[vr] = runsum[vr] * alpha[vr] + bs[vr];
#pragma unroll
    for (int nt = 0; nt < 4; ++nt)
#pragma unroll
      for (int vr = 0; vr < 8; ++vr)
        o[nt][vr] *= alpha[vr];

    // --- write P (bf16, C layout) to LDS so it can be re-read in A layout ---
#pragma unroll
    for (int vr = 0; vr < 8; ++vr) {
#pragma unroll
      for (int nt = 0; nt < 4; ++nt)
        Ps[vr + 8*hi][ln + 16*nt] = (bf16_t)s[nt][vr];
    }
    __syncthreads();   // Ps stores visible before fragment reads
#if USE_TDM
    __builtin_amdgcn_s_wait_tensorcnt(0);   // TDM finished writing Vs
#endif

    // --- O += P @ V : P is 16x64 (K=64), V block 64x64 -> four 16x16 tiles ---
    v16bf pf0 = ld_frag_2x16B(&Ps[ln][kb8],      &Ps[ln][kb8 + 16]);
    v16bf pf1 = ld_frag_2x16B(&Ps[ln][32 + kb8], &Ps[ln][32 + kb8 + 16]);
#pragma unroll
    for (int nt = 0; nt < 4; ++nt) {
      V16U vb0, vb1;
#pragma unroll
      for (int i = 0; i < 16; ++i) {
        vb0.b[i] = Vs[kb8 + i + (i >= 8 ? 8 : 0)][ln + 16*nt];        // B[k][n] = V[j0+k][d=n]
        vb1.b[i] = Vs[32 + kb8 + i + (i >= 8 ? 8 : 0)][ln + 16*nt];
      }
      o[nt] = __builtin_amdgcn_wmma_f32_16x16x32_bf16(
          false, pf0, false, vb0.v, (short)0, o[nt], false, false);
      o[nt] = __builtin_amdgcn_wmma_f32_16x16x32_bf16(
          false, pf1, false, vb1.v, (short)0, o[nt], false, false);
    }
    __syncthreads();   // Vs/Ps reads complete before next iteration's stores/TDM
  }

  // --- normalize and store (bf16 into attn buffer) ---
#pragma unroll
  for (int nt = 0; nt < 4; ++nt) {
#pragma unroll
    for (int vr = 0; vr < 8; ++vr) {
      const int irow = i0 + vr + 8*hi;
      O[qbase + (size_t)irow * DIMW + ln + 16*nt] =
          (bf16_t)(o[nt][vr] / runsum[vr]);
    }
  }
}

// ---------------------------------------------------------------------------
extern "C" void kernel_launch(void* const* d_in, const int* in_sizes, int n_in,
                              void* d_out, int out_size, void* d_ws, size_t ws_size,
                              hipStream_t stream) {
  const float* x  = (const float*)d_in[0];   // (B, M, DIM)
  const float* l  = (const float*)d_in[1];   // (B, N, DIM)
  const float* Wq = (const float*)d_in[2];
  const float* bq = (const float*)d_in[3];
  const float* Wk = (const float*)d_in[4];
  const float* bk = (const float*)d_in[5];
  const float* Wv = (const float*)d_in[6];
  const float* bv = (const float*)d_in[7];
  const float* Wo = (const float*)d_in[8];
  const float* bo = (const float*)d_in[9];
  float* out = (float*)d_out;

  const int B = 2;
  const size_t qElems  = (size_t)B * NLAT * DIMW;   // 2 Mi
  const size_t kvElems = (size_t)B * MTOK * DIMW;   // 8 Mi
  bf16_t* qb = (bf16_t*)d_ws;
  bf16_t* kb = qb + qElems;
  bf16_t* vb = kb + kvElems;
  bf16_t* ab = vb + kvElems;
  (void)in_sizes; (void)n_in; (void)out_size; (void)ws_size;

  const dim3 blk(256);
  // q = l @ Wq^T + bq   (2048 rows)
  gemm_bias_wmma<true, false><<<dim3(2048/256, DIMW/64), blk, 0, stream>>>(
      l, nullptr, Wq, bq, nullptr, qb);
  // k = x @ Wk^T + bk   (8192 rows)
  gemm_bias_wmma<true, false><<<dim3(8192/256, DIMW/64), blk, 0, stream>>>(
      x, nullptr, Wk, bk, nullptr, kb);
  // v = x @ Wv^T + bv
  gemm_bias_wmma<true, false><<<dim3(8192/256, DIMW/64), blk, 0, stream>>>(
      x, nullptr, Wv, bv, nullptr, vb);
  // flash attention: grid = (N/16, H, B), one wave per block
  attn_flash_wmma<<<dim3(NLAT/16, NHEADS, B), dim3(32), 0, stream>>>(qb, kb, vb, ab);
  // out = attn @ Wo^T + bo  (fp32 output)
  gemm_bias_wmma<false, true><<<dim3(2048/256, DIMW/64), blk, 0, stream>>>(
      nullptr, ab, Wo, bo, out, nullptr);
}
